// atentionHead_326417514636
// MI455X (gfx1250) — matrix operand order repaired and checked by reading, not compiled, not run on hardware
//
#include <hip/hip_runtime.h>
#include <hip/hip_bf16.h>

typedef __attribute__((ext_vector_type(16))) _Float16 v16h;
typedef __attribute__((ext_vector_type(8)))  _Float16 v8h;
typedef __attribute__((ext_vector_type(8)))  float    v8f;

union V16 { v16h v; v8h h[2]; };

#define CDIM 1024
#define DDIM 64
#define QT 64
#define KT 32
#define WPITCH 40   // f16 pitch of LDS weight tile rows (80B, 16B-aligned, conflict-free)
#define KPITCH 72   // f16 pitch of LDS K tile rows (144B)
#define VPITCH 40   // f16 pitch of LDS V^T tile rows (80B)
#define PPITCH 24   // f16 pitch of transposed P patch columns (48B, 16B-aligned)
// fold 1/sqrt(64) and log2(e) into Q so softmax can use exp2 directly
#define QSCALE (0.125f * 1.4426950408889634f)

// A-fragment (16-bit, 16xK=32) K index for element i of the lane's v16h
__device__ __forceinline__ int a_kk(int i, int half) {
    int vg = i >> 1, od = i & 1;
    return (vg < 4) ? (8 * half + 2 * vg + od)
                    : (16 + 8 * half + 2 * (vg - 4) + od);
}

// 32B LDS fragment load as two aligned 16B pieces
__device__ __forceinline__ V16 ldsFrag(const _Float16* p) {
    V16 r;
    r.h[0] = *(const v8h*)p;
    r.h[1] = *(const v8h*)(p + 8);
    return r;
}

// ---------------- kernel 1: convert weights fp32 -> f16 ----------------
__global__ void ah_prep_weights(const float* __restrict__ Wq,
                                const float* __restrict__ Wk,
                                const float* __restrict__ Wv,
                                _Float16* __restrict__ Wh) {
    int i = blockIdx.x * blockDim.x + threadIdx.x;
    const int per = DDIM * CDIM;           // 65536
    if (i >= 3 * per) return;
    const float* src = (i < per) ? Wq : (i < 2 * per ? Wk : Wv);
    Wh[i] = (_Float16)src[i % per];
}

// ---------------- kernel 2: fused QKV projection -----------------------
// block: 256 threads = 8 waves; wave owns 16 tokens x 192 channels.
// Weight tiles staged through double-buffered LDS; global loads for tile k+1
// issued before the 12 WMMAs of tile k; B-fragments rotated 2 deep.
__global__ __launch_bounds__(256) void ah_qkv_proj(
        const float* __restrict__ x, const _Float16* __restrict__ Wh,
        const float* __restrict__ bq, const float* __restrict__ bk,
        const float* __restrict__ bv,
        _Float16* __restrict__ Qh, _Float16* __restrict__ Kh,
        _Float16* __restrict__ Vt, int Tlen) {
    __shared__ _Float16 sW[2][192 * WPITCH];

    const int tid  = threadIdx.x;
    const int wave = tid >> 5;
    const int lane = tid & 31;
    const int col  = lane & 15;
    const int half = lane >> 4;
    const int m0   = blockIdx.x * 128 + wave * 16;

    // weight tile staging: 768 16B-units, 3 per thread
    auto loadW = [&](int c0, v8h* g) {
        #pragma unroll
        for (int j = 0; j < 3; ++j) {
            int u = tid + 256 * j, row = u >> 2, part = u & 3;
            g[j] = *(const v8h*)(Wh + (size_t)row * CDIM + c0 + part * 8);
        }
    };
    auto storeW = [&](int bufi, const v8h* g) {
        #pragma unroll
        for (int j = 0; j < 3; ++j) {
            int u = tid + 256 * j, row = u >> 2, part = u & 3;
            *(v8h*)&sW[bufi][row * WPITCH + part * 8] = g[j];
        }
    };

    const float* xr = x + (size_t)(m0 + col) * CDIM;
    auto loadA = [&](int c0, v8f& lo, v8f& hi) {
        lo = *(const v8f*)(xr + c0 + 16 * half);       // covers kk = 16h..16h+7
        hi = *(const v8f*)(xr + c0 + 16 * half + 8);   // covers kk = 16h+8..+15
    };

    v8f acc[12];
    #pragma unroll
    for (int nt = 0; nt < 12; ++nt) acc[nt] = {};

    v8h wreg[3];
    v8f alo, ahi;
    loadW(0, wreg);
    storeW(0, wreg);
    loadA(0, alo, ahi);

    for (int kb = 0; kb < CDIM / 32; ++kb) {
        const int buf = kb & 1;
        v8h wnext[3];
        v8f nlo, nhi;
        if (kb < CDIM / 32 - 1) {          // issue next tile's global loads early
            loadW((kb + 1) * 32, wnext);
            loadA((kb + 1) * 32, nlo, nhi);
        }
        v16h a;
        #pragma unroll
        for (int i = 0; i < 16; ++i)
            a[i] = (_Float16)(i < 8 ? alo[i] : ahi[i - 8]);

        __syncthreads();                   // tile `buf` visible; prev reads done

        // B fragments rotated 2 deep so ds_load(nt+2) issues before wmma(nt)
        V16 bfr[2];
        bfr[0] = ldsFrag(&sW[buf][(col)      * WPITCH + 16 * half]);
        bfr[1] = ldsFrag(&sW[buf][(16 + col) * WPITCH + 16 * half]);
        #pragma unroll
        for (int nt = 0; nt < 12; ++nt) {
            V16 cur = bfr[nt & 1];
            if (nt + 2 < 12)
                bfr[nt & 1] = ldsFrag(&sW[buf][((nt + 2) * 16 + col) * WPITCH + 16 * half]);
            acc[nt] = __builtin_amdgcn_wmma_f32_16x16x32_f16(
                false, a, false, cur.v, (short)0, acc[nt], false, false);
        }
        if (kb < CDIM / 32 - 1) {
            storeW(buf ^ 1, wnext);
            alo = nlo; ahi = nhi;
        }
    }

    #pragma unroll
    for (int nt = 0; nt < 12; ++nt) {
        int chn = nt * 16 + col;
        int p = chn >> 6, dch = chn & 63;
        float bias = (p == 0 ? bq : (p == 1 ? bk : bv))[dch];
        #pragma unroll
        for (int v = 0; v < 8; ++v) {
            int t = m0 + v + 8 * half;
            float val = acc[nt][v] + bias;
            if (p == 0) {
                Qh[(size_t)t * DDIM + dch] = (_Float16)(val * QSCALE);
            } else if (p == 1) {
                Kh[(size_t)t * DDIM + dch] = (_Float16)val;
            } else {
                int b = t / Tlen, tt = t % Tlen;       // V stored transposed [B][d][T]
                Vt[(size_t)(b * DDIM + dch) * Tlen + tt] = (_Float16)val;
            }
        }
    }
}

// ---------------- kernel 3: causal flash attention ---------------------
// block: 128 threads = 4 waves; wave owns 16 query rows; K/V 32-key tiles
// double-buffered in LDS. Row-sums of P computed by an extra WMMA against an
// all-ones B fragment (oL), eliminating the sum shuffles and l bookkeeping.
__global__ __launch_bounds__(128) void ah_attn(
        const _Float16* __restrict__ Qh, const _Float16* __restrict__ Kh,
        const _Float16* __restrict__ Vt, float* __restrict__ out, int Tlen) {
    __shared__ _Float16 sK[2][32 * KPITCH];
    __shared__ _Float16 sV[2][64 * VPITCH];
    __shared__ _Float16 sP[4 * 32 * PPITCH];   // transposed P patch per wave

    const int tid  = threadIdx.x;
    const int wave = tid >> 5, lane = tid & 31;
    const int col  = lane & 15, half = lane >> 4;
    const int tilesPerBatch = Tlen / QT;
    const int b  = blockIdx.x / tilesPerBatch;
    const int qt = (blockIdx.x % tilesPerBatch) * QT;
    const int q0 = qt + wave * 16;
    const size_t bT = (size_t)b * Tlen;

    // K tile: 256 16B-units (row=u>>3, part=u&7); V tile: 256 units (row=u>>2)
    auto loadKV = [&](int k0, v8h* g) {
        #pragma unroll
        for (int j = 0; j < 2; ++j) {
            int u = tid + 128 * j, row = u >> 3, part = u & 7;
            g[j] = *(const v8h*)(Kh + (bT + k0 + row) * DDIM + part * 8);
        }
        #pragma unroll
        for (int j = 0; j < 2; ++j) {
            int u = tid + 128 * j, row = u >> 2, part = u & 3;
            g[2 + j] = *(const v8h*)(Vt + (size_t)(b * DDIM + row) * Tlen + k0 + part * 8);
        }
    };
    auto storeKV = [&](int bufi, const v8h* g) {
        #pragma unroll
        for (int j = 0; j < 2; ++j) {
            int u = tid + 128 * j, row = u >> 3, part = u & 7;
            *(v8h*)&sK[bufi][row * KPITCH + part * 8] = g[j];
        }
        #pragma unroll
        for (int j = 0; j < 2; ++j) {
            int u = tid + 128 * j, row = u >> 2, part = u & 3;
            *(v8h*)&sV[bufi][row * VPITCH + part * 8] = g[2 + j];
        }
    };

    // Q A-fragments (d = 0..31, 32..63), resident for whole kernel
    v16h qf[2];
    {
        const _Float16* qr = Qh + (bT + q0 + col) * DDIM;
        #pragma unroll
        for (int cc = 0; cc < 2; ++cc)
            #pragma unroll
            for (int i = 0; i < 16; ++i)
                qf[cc][i] = qr[cc * 32 + 16 * half + i];
    }

    // all-ones B fragment: oL += P x ones gives every lane its row's sum(exp)
    v16h ones;
    #pragma unroll
    for (int i = 0; i < 16; ++i) ones[i] = (_Float16)1.0f;

    v8f o0 = {}, o1 = {}, o2 = {}, o3 = {}, oL = {};
    float mrow[8];
    #pragma unroll
    for (int v = 0; v < 8; ++v) mrow[v] = -__builtin_inff();

    const int nIter = (qt + QT) / KT;
    v8h kv[4];
    loadKV(0, kv);
    storeKV(0, kv);

    for (int it = 0; it < nIter; ++it) {
        const int k0 = it * KT;
        const int buf = it & 1;
        v8h kvn[4];
        if (it + 1 < nIter) loadKV(k0 + KT, kvn);   // issue global loads early

        __syncthreads();

        // all 4 K fragments first (one clause), then 4 back-to-back WMMAs
        V16 kb0 = ldsFrag(&sK[buf][(col)      * KPITCH + 16 * half]);
        V16 kb1 = ldsFrag(&sK[buf][(16 + col) * KPITCH + 16 * half]);
        V16 kb2 = ldsFrag(&sK[buf][(col)      * KPITCH + 32 + 16 * half]);
        V16 kb3 = ldsFrag(&sK[buf][(16 + col) * KPITCH + 32 + 16 * half]);
        v8f s0 = {}, s1 = {};
        s0 = __builtin_amdgcn_wmma_f32_16x16x32_f16(false, qf[0], false, kb0.v,
                                                    (short)0, s0, false, false);
        s1 = __builtin_amdgcn_wmma_f32_16x16x32_f16(false, qf[0], false, kb1.v,
                                                    (short)0, s1, false, false);
        s0 = __builtin_amdgcn_wmma_f32_16x16x32_f16(false, qf[1], false, kb2.v,
                                                    (short)0, s0, false, false);
        s1 = __builtin_amdgcn_wmma_f32_16x16x32_f16(false, qf[1], false, kb3.v,
                                                    (short)0, s1, false, false);

        // issue V-fragment loads now; latency hides under the softmax VALU
        V16 vb0 = ldsFrag(&sV[buf][(col)      * VPITCH + 16 * half]);
        V16 vb1 = ldsFrag(&sV[buf][(16 + col) * VPITCH + 16 * half]);
        V16 vb2 = ldsFrag(&sV[buf][(32 + col) * VPITCH + 16 * half]);
        V16 vb3 = ldsFrag(&sV[buf][(48 + col) * VPITCH + 16 * half]);

        // causal mask + online softmax (row = v + 8*half, key col = lane%16)
        v8h pk0, pk1;
        #pragma unroll
        for (int v = 0; v < 8; ++v) {
            int qpos = q0 + v + 8 * half;
            float a0 = (k0 + col      > qpos) ? -__builtin_inff() : s0[v];
            float a1 = (k0 + 16 + col > qpos) ? -__builtin_inff() : s1[v];
            float rm = fmaxf(a0, a1);
            rm = fmaxf(rm, __shfl_xor(rm, 1, 32));
            rm = fmaxf(rm, __shfl_xor(rm, 2, 32));
            rm = fmaxf(rm, __shfl_xor(rm, 4, 32));
            rm = fmaxf(rm, __shfl_xor(rm, 8, 32));
            float mnew = fmaxf(mrow[v], rm);
            float alpha = exp2f(mrow[v] - mnew);
            mrow[v] = mnew;
            o0[v] *= alpha; o1[v] *= alpha; o2[v] *= alpha; o3[v] *= alpha;
            oL[v] *= alpha;
            pk0[v] = (_Float16)exp2f(a0 - mnew);
            pk1[v] = (_Float16)exp2f(a1 - mnew);
        }

        // C-layout P -> A-layout via transposed per-wave LDS patch:
        // addr = kk*PPITCH + row; each frag stored as one packed b128
        _Float16* pw = &sP[wave * 32 * PPITCH];
        *(v8h*)&pw[(col)      * PPITCH + 8 * half] = pk0;   // kk = col
        *(v8h*)&pw[(16 + col) * PPITCH + 8 * half] = pk1;   // kk = 16+col
        asm volatile("s_wait_dscnt 0" ::: "memory");
        v16h pa;
        #pragma unroll
        for (int i = 0; i < 16; ++i)
            pa[i] = pw[(16 * half + i) * PPITCH + col];

        // O += P * V (4 d-tiles) and row-sum accumulate via ones fragment
        o0 = __builtin_amdgcn_wmma_f32_16x16x32_f16(false, pa, false, vb0.v,
                                                    (short)0, o0, false, false);
        o1 = __builtin_amdgcn_wmma_f32_16x16x32_f16(false, pa, false, vb1.v,
                                                    (short)0, o1, false, false);
        o2 = __builtin_amdgcn_wmma_f32_16x16x32_f16(false, pa, false, vb2.v,
                                                    (short)0, o2, false, false);
        o3 = __builtin_amdgcn_wmma_f32_16x16x32_f16(false, pa, false, vb3.v,
                                                    (short)0, o3, false, false);
        oL = __builtin_amdgcn_wmma_f32_16x16x32_f16(false, pa, false, ones,
                                                    (short)0, oL, false, false);

        if (it + 1 < nIter) storeKV(buf ^ 1, kvn);  // store late into other buffer
    }

    float* orow = out + (bT + q0) * DDIM;
    #pragma unroll
    for (int v = 0; v < 8; ++v) {
        int r = v + 8 * half;
        float inv = 1.0f / oL[v];
        orow[(size_t)r * DDIM + col]      = o0[v] * inv;
        orow[(size_t)r * DDIM + 16 + col] = o1[v] * inv;
        orow[(size_t)r * DDIM + 32 + col] = o2[v] * inv;
        orow[(size_t)r * DDIM + 48 + col] = o3[v] * inv;
    }
}

extern "C" void kernel_launch(void* const* d_in, const int* in_sizes, int n_in,
                              void* d_out, int out_size, void* d_ws, size_t ws_size,
                              hipStream_t stream) {
    const float* x  = (const float*)d_in[0];
    const float* Wq = (const float*)d_in[1];
    const float* bq = (const float*)d_in[2];
    const float* Wk = (const float*)d_in[3];
    const float* bk = (const float*)d_in[4];
    const float* Wv = (const float*)d_in[5];
    const float* bv = (const float*)d_in[6];
    float* out = (float*)d_out;

    const int N = in_sizes[0] / CDIM;   // B*T = 16384
    const int T = 4096;
    const int B = N / T;                // 4

    _Float16* Wh = (_Float16*)d_ws;               // 3*64*1024 f16
    _Float16* Qh = Wh + 3 * DDIM * CDIM;          // [B*T][64]
    _Float16* Kh = Qh + (size_t)N * DDIM;         // [B*T][64]
    _Float16* Vt = Kh + (size_t)N * DDIM;         // [B][64][T]

    {   // weights -> f16
        int n = 3 * DDIM * CDIM;
        ah_prep_weights<<<(n + 255) / 256, 256, 0, stream>>>(Wq, Wk, Wv, Wh);
    }
    {   // fused QKV projection: 128 rows per block
        ah_qkv_proj<<<N / 128, 256, 0, stream>>>(x, Wh, bq, bk, bv, Qh, Kh, Vt, T);
    }
    {   // causal flash attention: one block per 64 query rows
        ah_attn<<<B * (T / QT), 128, 0, stream>>>(Qh, Kh, Vt, out, T);
    }
}
